// GridSelfAttention_65034394796582
// MI455X (gfx1250) — matrix operand order repaired and checked by reading, not compiled, not run on hardware
//
#include <hip/hip_runtime.h>
#include <hip/hip_bf16.h>

#define NTOK   384
#define CPAIR  128
#define NHEAD  4
#define HDIM   32
#define NNROWS (NTOK * NTOK)      // 147456 rows of 128 channels

typedef __bf16 bf16;
typedef __attribute__((ext_vector_type(16))) __bf16 v16bf;
typedef __attribute__((ext_vector_type(8)))  __bf16 v8bf;
typedef __attribute__((ext_vector_type(4)))  __bf16 v4bf;
typedef __attribute__((ext_vector_type(8)))  float  v8f;

union U16 { v16bf v; v8bf h[2]; };

__device__ __forceinline__ v8f wmma_bf16(v16bf a, v16bf b, v8f c) {
    // D = A(16x32 bf16) * B(32x16 bf16) + C(16x16 f32)
    return __builtin_amdgcn_wmma_f32_16x16x32_bf16(
        /*neg_a=*/false, a, /*neg_b=*/false, b,
        /*c_mod=*/(short)0, c, /*reuse_a=*/false, /*reuse_b=*/false);
}

// ---------------------------------------------------------------------------
// Weight f32 -> bf16 conversion (tiny)
// ---------------------------------------------------------------------------
__global__ void cvt_w_kernel(const float* __restrict__ src, bf16* __restrict__ dst, int n) {
    int i = blockIdx.x * 256 + threadIdx.x;
    if (i < n) dst[i] = (bf16)src[i];
}

// ---------------------------------------------------------------------------
// LayerNorm over C=128; one wave per row, 4 channels per lane. Output bf16.
// ---------------------------------------------------------------------------
__global__ void ln_kernel(const float* __restrict__ pair,
                          const float* __restrict__ lsc,
                          const float* __restrict__ lbi,
                          bf16* __restrict__ xb) {
    const int wv = threadIdx.x >> 5;
    const int lane = threadIdx.x & 31;
    const size_t row = (size_t)blockIdx.x * 8 + wv;
    const float4 x = *(const float4*)(pair + row * CPAIR + lane * 4);

    float s  = x.x + x.y + x.z + x.w;
    float s2 = x.x * x.x + x.y * x.y + x.z * x.z + x.w * x.w;
    #pragma unroll
    for (int o = 16; o > 0; o >>= 1) {
        s  += __shfl_xor(s,  o, 32);
        s2 += __shfl_xor(s2, o, 32);
    }
    const float mu = s * (1.0f / CPAIR);
    const float var = s2 * (1.0f / CPAIR) - mu * mu;
    const float rs = rsqrtf(var + 1e-5f);

    const float4 sc = *(const float4*)(lsc + lane * 4);
    const float4 bi = *(const float4*)(lbi + lane * 4);
    v4bf y;
    y[0] = (bf16)((x.x - mu) * rs * sc.x + bi.x);
    y[1] = (bf16)((x.y - mu) * rs * sc.y + bi.y);
    y[2] = (bf16)((x.z - mu) * rs * sc.z + bi.z);
    y[3] = (bf16)((x.w - mu) * rs * sc.w + bi.w);
    *(v4bf*)(xb + row * CPAIR + lane * 4) = y;
}

// ---------------------------------------------------------------------------
// bias planes: biasb[h][q*384+k] = dot(x_row, w_bias[h]); one wave per row.
// ---------------------------------------------------------------------------
__global__ void bias_kernel(const bf16* __restrict__ xb,
                            const float* __restrict__ w_bias,
                            float* __restrict__ biasb) {
    const int wv = threadIdx.x >> 5;
    const int lane = threadIdx.x & 31;
    const size_t row = (size_t)blockIdx.x * 8 + wv;

    const v4bf xv = *(const v4bf*)(xb + row * CPAIR + lane * 4);
    const float x0 = (float)xv[0], x1 = (float)xv[1], x2 = (float)xv[2], x3 = (float)xv[3];

    float acc[NHEAD];
    #pragma unroll
    for (int h = 0; h < NHEAD; ++h) {
        const float4 w = *(const float4*)(w_bias + h * CPAIR + lane * 4);
        float a = x0 * w.x + x1 * w.y + x2 * w.z + x3 * w.w;
        #pragma unroll
        for (int o = 16; o > 0; o >>= 1) a += __shfl_xor(a, o, 32);
        acc[h] = a;
    }
    if (lane == 0) {
        #pragma unroll
        for (int h = 0; h < NHEAD; ++h)
            biasb[(size_t)h * NNROWS + row] = acc[h];
    }
}

// ---------------------------------------------------------------------------
// GEMM: Y[M,128] = X[M,128] * W^T, W is [128 out][128 in] bf16.
// 256 threads (8 waves); each wave owns a 32(M)x128(N) strip (2 A-tiles,
// 64 WMMAs), K in 4 steps with A-tiles preloaded one step ahead.
// W staged in LDS, pitch 136 halves.
// MODE 0: bf16 row-major (scaled).  MODE 1: bf16 scatter into vT layout.
// MODE 2: f32 row-major.
// ---------------------------------------------------------------------------
template <int MODE>
__global__ void gemm_x128(const bf16* __restrict__ X, const bf16* __restrict__ W,
                          void* __restrict__ Yv, float scale) {
    __shared__ bf16 wl[128 * 136];
    const int tid = threadIdx.x;
    const int lane = tid & 31;
    const int wv = tid >> 5;
    const int lm = lane & 15, lh = lane >> 4;

    for (int i = tid; i < 2048; i += 256) {
        const int r = i >> 4, c = (i & 15) * 8;
        *(v8bf*)&wl[r * 136 + c] = *(const v8bf*)&W[r * 128 + c];
    }

    const size_t rowbase = (size_t)blockIdx.x * 256 + wv * 32;
    const bf16* abase = X + (rowbase + lm) * CPAIR + lh * 8;

    U16 a0, a1;   // current k-step A-tiles (preloaded)
    a0.h[0] = *(const v8bf*)abase;
    a0.h[1] = *(const v8bf*)(abase + 16);
    a1.h[0] = *(const v8bf*)(abase + 16 * CPAIR);
    a1.h[1] = *(const v8bf*)(abase + 16 * CPAIR + 16);

    __syncthreads();

    v8f acc[2][8] = {};
    #pragma unroll
    for (int ks = 0; ks < 4; ++ks) {
        const U16 c0 = a0, c1 = a1;
        if (ks < 3) {
            const bf16* ap = abase + (ks + 1) * 32;
            a0.h[0] = *(const v8bf*)ap;
            a0.h[1] = *(const v8bf*)(ap + 16);
            a1.h[0] = *(const v8bf*)(ap + 16 * CPAIR);
            a1.h[1] = *(const v8bf*)(ap + 16 * CPAIR + 16);
        }
        #pragma unroll
        for (int nt = 0; nt < 8; ++nt) {
            U16 b;
            const bf16* bp = &wl[(nt * 16 + lm) * 136 + ks * 32 + lh * 8];
            b.h[0] = *(const v8bf*)bp;
            b.h[1] = *(const v8bf*)(bp + 16);
            acc[0][nt] = wmma_bf16(c0.v, b.v, acc[0][nt]);
            acc[1][nt] = wmma_bf16(c1.v, b.v, acc[1][nt]);
        }
    }

    #pragma unroll
    for (int t = 0; t < 2; ++t) {
        if (MODE == 0) {
            bf16* Y = (bf16*)Yv;
            #pragma unroll
            for (int nt = 0; nt < 8; ++nt)
                #pragma unroll
                for (int r = 0; r < 8; ++r) {
                    const size_t row = rowbase + t * 16 + r + 8 * lh;
                    Y[row * CPAIR + nt * 16 + lm] = (bf16)(acc[t][nt][r] * scale);
                }
        } else if (MODE == 1) {
            bf16* Y = (bf16*)Yv;   // vT[(b*128 + c)*384 + kidx]
            #pragma unroll
            for (int nt = 0; nt < 8; ++nt)
                #pragma unroll
                for (int r = 0; r < 8; ++r) {
                    const unsigned row = (unsigned)(rowbase + t * 16 + r + 8 * lh);
                    const unsigned bb = row / NTOK;
                    const unsigned kidx = row - bb * NTOK;
                    const unsigned c = nt * 16 + lm;
                    Y[((size_t)bb * CPAIR + c) * NTOK + kidx] = (bf16)acc[t][nt][r];
                }
        } else {
            float* Y = (float*)Yv;
            #pragma unroll
            for (int nt = 0; nt < 8; ++nt)
                #pragma unroll
                for (int r = 0; r < 8; ++r) {
                    const size_t row = rowbase + t * 16 + r + 8 * lh;
                    Y[row * CPAIR + nt * 16 + lm] = acc[t][nt][r];
                }
        }
    }
}

// ---------------------------------------------------------------------------
// Attention: one 2-wave block per (qtile16, head, row-batch b).
// Phase0: prefill S-LDS with bias plane (coalesced) + mask (-1e9 fold).
// Phase1: S = Q K^T + C(bias from LDS); K tiles software-pipelined.
// Phase2: softmax, float4-vectorized LDS traffic, P -> LDS bf16.
// Phase3: O = P V; V tiles pipelined, dual accumulators to break the
//         WMMA RAW hazard chain; fuse sigmoid gate, write bf16.
// ---------------------------------------------------------------------------
__global__ void attn_kernel(const bf16* __restrict__ qb, const bf16* __restrict__ kb,
                            const bf16* __restrict__ vT, const bf16* __restrict__ glog,
                            const float* __restrict__ biasb, const int* __restrict__ mask,
                            bf16* __restrict__ og) {
    __shared__ float Sls[16 * 388];
    __shared__ bf16  Pls[16 * 392];

    const int lane = threadIdx.x & 31;
    const int wv = threadIdx.x >> 5;
    const int qt = blockIdx.x;   // 0..23
    const int h  = blockIdx.y;   // 0..3
    const int b  = blockIdx.z;   // 0..383
    const int lm = lane & 15, lh = lane >> 4;

    // Phase 0: prefill S with bias + mask (coalesced float4/int4 loads)
    {
        const int trow = threadIdx.x >> 2;   // 0..15
        const int tsub = threadIdx.x & 3;
        const float* bplane = biasb + (size_t)h * NNROWS + (size_t)(qt * 16 + trow) * NTOK;
        const int* mrow = mask + b * NTOK;
        #pragma unroll
        for (int j = 0; j < 24; ++j) {
            const int c4 = (tsub * 24 + j) * 4;
            const float4 bv = *(const float4*)(bplane + c4);
            const int4 mv = *(const int4*)(mrow + c4);
            float4 o;
            o.x = mv.x ? bv.x : -1e9f;
            o.y = mv.y ? bv.y : -1e9f;
            o.z = mv.z ? bv.z : -1e9f;
            o.w = mv.w ? bv.w : -1e9f;
            *(float4*)&Sls[trow * 388 + c4] = o;
        }
    }

    // Q tile into A-layout registers (held across phase 1)
    U16 a;
    {
        const bf16* qp = qb + ((size_t)(b * NTOK + qt * 16 + lm)) * CPAIR + h * HDIM + lh * 8;
        a.h[0] = *(const v8bf*)qp;
        a.h[1] = *(const v8bf*)(qp + 16);
    }
    __syncthreads();

    // Phase 1: logits; K tiles preloaded one tile ahead, bias via C operand
    {
        const int kt0 = wv * 12;
        const bf16* kcol = kb + ((size_t)(b * NTOK + lm)) * CPAIR + h * HDIM + lh * 8;
        U16 bm;
        {
            const bf16* kp = kcol + (size_t)(kt0 * 16) * CPAIR;
            bm.h[0] = *(const v8bf*)kp;
            bm.h[1] = *(const v8bf*)(kp + 16);
        }
        #pragma unroll
        for (int i = 0; i < 12; ++i) {
            const U16 cur = bm;
            if (i < 11) {
                const bf16* kp = kcol + (size_t)((kt0 + i + 1) * 16) * CPAIR;
                __builtin_prefetch(kp + 16 * CPAIR, 0, 3);   // tile i+2
                bm.h[0] = *(const v8bf*)kp;
                bm.h[1] = *(const v8bf*)(kp + 16);
            }
            const int kg = (kt0 + i) * 16 + lm;
            v8f c;
            #pragma unroll
            for (int r = 0; r < 8; ++r) c[r] = Sls[(r + 8 * lh) * 388 + kg];
            c = wmma_bf16(a.v, cur.v, c);
            #pragma unroll
            for (int r = 0; r < 8; ++r) Sls[(r + 8 * lh) * 388 + kg] = c[r];
        }
    }
    __syncthreads();

    // Phase 2: softmax over k (16 rows x 384; 4 threads per row, float4 ops)
    {
        const int row = threadIdx.x >> 2;
        const int sub = threadIdx.x & 3;
        float* srow = &Sls[row * 388 + sub * 96];
        float mx = -3.4e38f;
        #pragma unroll
        for (int j = 0; j < 24; ++j) {
            const float4 v = *(const float4*)(srow + j * 4);
            mx = fmaxf(mx, fmaxf(fmaxf(v.x, v.y), fmaxf(v.z, v.w)));
        }
        mx = fmaxf(mx, __shfl_xor(mx, 1, 32));
        mx = fmaxf(mx, __shfl_xor(mx, 2, 32));
        float sum = 0.0f;
        #pragma unroll
        for (int j = 0; j < 24; ++j) {
            float4 v = *(const float4*)(srow + j * 4);
            v.x = __expf(v.x - mx);
            v.y = __expf(v.y - mx);
            v.z = __expf(v.z - mx);
            v.w = __expf(v.w - mx);
            *(float4*)(srow + j * 4) = v;
            sum += (v.x + v.y) + (v.z + v.w);
        }
        sum += __shfl_xor(sum, 1, 32);
        sum += __shfl_xor(sum, 2, 32);
        const float inv = 1.0f / sum;
        bf16* prow = &Pls[row * 392 + sub * 96];
        #pragma unroll
        for (int j = 0; j < 24; ++j) {
            const float4 v = *(const float4*)(srow + j * 4);
            v4bf p;
            p[0] = (bf16)(v.x * inv);
            p[1] = (bf16)(v.y * inv);
            p[2] = (bf16)(v.z * inv);
            p[3] = (bf16)(v.w * inv);
            *(v4bf*)(prow + j * 4) = p;
        }
    }
    __syncthreads();

    // Phase 3: O = P V; wave wv owns d-tile wv; dual accumulators, V pipelined
    v8f o0 = {}, o1 = {};
    const int d0 = wv * 16;
    const bf16* vrow = vT + ((size_t)((b * NHEAD + h) * HDIM + d0 + lm)) * NTOK;
    U16 vb;
    vb.h[0] = *(const v8bf*)(vrow + lh * 8);
    vb.h[1] = *(const v8bf*)(vrow + lh * 8 + 16);
    #pragma unroll
    for (int ks = 0; ks < 12; ++ks) {
        const U16 vcur = vb;
        if (ks < 11) {
            const bf16* vp = vrow + (ks + 1) * 32 + lh * 8;
            vb.h[0] = *(const v8bf*)vp;
            vb.h[1] = *(const v8bf*)(vp + 16);
        }
        U16 pa;
        const bf16* pp = &Pls[lm * 392 + ks * 32 + lh * 8];
        pa.h[0] = *(const v8bf*)pp;
        pa.h[1] = *(const v8bf*)(pp + 16);
        if (ks & 1) o1 = wmma_bf16(pa.v, vcur.v, o1);
        else        o0 = wmma_bf16(pa.v, vcur.v, o0);
    }
    const v8f o = o0 + o1;

    #pragma unroll
    for (int r = 0; r < 8; ++r) {
        const int qg = qt * 16 + r + 8 * lh;
        const size_t idx = ((size_t)(b * NTOK + qg)) * CPAIR + h * HDIM + d0 + lm;
        float g = (float)glog[idx];
        g = 1.0f / (1.0f + __expf(-g));
        og[idx] = (bf16)(o[r] * g);
    }
}

// ---------------------------------------------------------------------------
extern "C" void kernel_launch(void* const* d_in, const int* in_sizes, int n_in,
                              void* d_out, int out_size, void* d_ws, size_t ws_size,
                              hipStream_t stream) {
    const float* pair     = (const float*)d_in[0];
    const int*   mask     = (const int*)d_in[1];
    const float* ln_scale = (const float*)d_in[2];
    const float* ln_bias  = (const float*)d_in[3];
    const float* w_bias   = (const float*)d_in[4];
    const float* w_q      = (const float*)d_in[5];
    const float* w_k      = (const float*)d_in[6];
    const float* w_v      = (const float*)d_in[7];
    const float* w_gate   = (const float*)d_in[8];
    const float* w_out    = (const float*)d_in[9];
    (void)in_sizes; (void)n_in; (void)out_size; (void)ws_size;

    const size_t NNC = (size_t)NNROWS * CPAIR;
    bf16* xb  = (bf16*)d_ws;          // LN output; later reused as gated o
    bf16* qb  = xb + NNC;
    bf16* kb  = qb + NNC;
    bf16* vT  = kb + NNC;
    bf16* gl  = vT + NNC;
    bf16* wqb = gl + NNC;
    bf16* wkb = wqb + CPAIR * CPAIR;
    bf16* wvb = wkb + CPAIR * CPAIR;
    bf16* wgb = wvb + CPAIR * CPAIR;
    bf16* wob = wgb + CPAIR * CPAIR;
    float* biasb = (float*)(wob + CPAIR * CPAIR);   // 4 planes of [N*N] f32
    bf16* og = xb;

    cvt_w_kernel<<<64, 256, 0, stream>>>(w_q,    wqb, CPAIR * CPAIR);
    cvt_w_kernel<<<64, 256, 0, stream>>>(w_k,    wkb, CPAIR * CPAIR);
    cvt_w_kernel<<<64, 256, 0, stream>>>(w_v,    wvb, CPAIR * CPAIR);
    cvt_w_kernel<<<64, 256, 0, stream>>>(w_gate, wgb, CPAIR * CPAIR);
    cvt_w_kernel<<<64, 256, 0, stream>>>(w_out,  wob, CPAIR * CPAIR);

    ln_kernel<<<NNROWS / 8, 256, 0, stream>>>(pair, ln_scale, ln_bias, xb);
    bias_kernel<<<NNROWS / 8, 256, 0, stream>>>(xb, w_bias, biasb);

    const float qscale = 0.17677669529663687f;  // 32^-0.5
    gemm_x128<0><<<NNROWS / 256, 256, 0, stream>>>(xb, wqb, qb, qscale);
    gemm_x128<0><<<NNROWS / 256, 256, 0, stream>>>(xb, wkb, kb, 1.0f);
    gemm_x128<1><<<NNROWS / 256, 256, 0, stream>>>(xb, wvb, vT, 1.0f);
    gemm_x128<0><<<NNROWS / 256, 256, 0, stream>>>(xb, wgb, gl, 1.0f);

    attn_kernel<<<dim3(NTOK / 16, NHEAD, NTOK), 64, 0, stream>>>(
        qb, kb, vT, gl, biasb, mask, og);

    gemm_x128<2><<<NNROWS / 256, 256, 0, stream>>>(og, wob, d_out, 1.0f);
}
